// ConditionalFeedForward_59399397704333
// MI455X (gfx1250) — compile-verified, matrix operand order checked by reading, not run on hardware
//
#include <hip/hip_runtime.h>

typedef __attribute__((ext_vector_type(16))) __bf16 v16bf;
typedef __attribute__((ext_vector_type(8)))  float  v8f;
typedef __attribute__((ext_vector_type(4)))  unsigned int v4u;
typedef __attribute__((ext_vector_type(8)))  int  v8i;
typedef __attribute__((ext_vector_type(4)))  int  v4i;

#define TT   1024
#define AA   2
#define EE   8
#define DD   2048
#define II   5632
#define MAXP (TT * AA)

// ---------------- routing ----------------

__global__ void k_zero(int* cnt) {
  if (threadIdx.x < EE) cnt[threadIdx.x] = 0;
}

__global__ void k_route(const int* __restrict__ idx, int* __restrict__ cnt,
                        int* __restrict__ list) {
  int p = blockIdx.x * blockDim.x + threadIdx.x;
  if (p >= MAXP) return;
  int e = idx[p];
  int pos = atomicAdd(&cnt[e], 1);
  list[e * MAXP + pos] = p;
}

__device__ __forceinline__ float fast_silu(float v) {
  // v * sigmoid(v) with v_rcp_f32 instead of the IEEE division chain
  return v * __builtin_amdgcn_rcpf(1.0f + __expf(-v));
}

// ---------------- up projection: h = silu(x W1^T) * (x W3^T) ----------------
// Block = 4 waves; each wave owns a 16(token) x 32(I) tile (two 16x16 WMMA tiles
// sharing the A operand -> 4 v_wmma per K-step). The gathered 16-token x tile
// (16 x DD f32) is staged into LDS once per block by a gather-mode TDM load.
// Grid: x = token-blocks (fast) so all blocks sharing one weight slice dispatch
// adjacently and the slice is HBM-read once, L2-served afterwards.
// CDNA5 16-bit A layout (16x32): lanes 0-15 row M=l&15, K = {2j,2j+1|j<4} U {16+..};
// lanes 16-31 K offset +8. B mirrors with N = l&15. C/D: VGPR r -> M=r+8*half.

__global__ void __launch_bounds__(128)
k_up(const float* __restrict__ x, const float* __restrict__ w1,
     const float* __restrict__ w3, const int* __restrict__ cnt,
     const int* __restrict__ list, __bf16* __restrict__ hbuf)
{
  extern __shared__ float xs[];                // 16 rows x DD f32 = 128 KB
  const int e    = blockIdx.z;
  const int n    = cnt[e];
  const int row0 = blockIdx.x * 16;            // token-block on fast grid dim
  if (row0 >= n) return;                       // uniform exit
  const int wave  = threadIdx.x >> 5;
  const int lane  = threadIdx.x & 31;
  const int half  = lane >> 4;
  const int mn    = lane & 15;
  const int ibase = (blockIdx.y * 8 + wave * 2) * 16;   // two i-tiles per wave

  if (wave == 0) {
    // Gather-mode Tensor-DMA: 16 token rows of x -> LDS, full D width.
    unsigned short tok[16];
#pragma unroll
    for (int r = 0; r < 16; ++r) {
      int rr = row0 + r; if (rr > n - 1) rr = n - 1;
      tok[r] = (unsigned short)(list[e * MAXP + rr] >> 1);   // token = pair / A
    }
    const unsigned long long ga = (unsigned long long)x;
    v4u g0;
    g0.x = 1u | (1u << 31);                       // count=1, gather_mode=1 (16-bit idx)
    g0.y = 0u;                                    // lds_addr (base of dynamic LDS)
    g0.z = (unsigned int)(ga & 0xffffffffu);      // global_addr[31:0]
    g0.w = (unsigned int)((ga >> 32) & 0x01ffffffu) | (2u << 30); // addr[56:32] | type=2
    v8i g1;
    g1[0] = 0x00020000;                           // wg_mask=0, data_size=2 (4 bytes)
    g1[1] = (int)(((unsigned)DD & 0xffffu) << 16);// tensor_dim0 lo16 (bits 63:48)
    g1[2] = (int)(((unsigned)TT & 0xffffu) << 16);// dim0 hi=0 | tensor_dim1 lo16
    g1[3] = (int)(((unsigned)DD & 0xffffu) << 16);// dim1 hi=0 | tile_dim0 = DD
    g1[4] = 16;                                   // tile_dim1 = #valid row indices
    g1[5] = DD;                                   // tensor_dim0_stride lo32
    g1[6] = 0;
    g1[7] = 0;
    v4i g2, g3;
#pragma unroll
    for (int r = 0; r < 4; ++r) {
      g2[r] = (int)((unsigned)tok[2 * r]     | ((unsigned)tok[2 * r + 1] << 16));
      g3[r] = (int)((unsigned)tok[8 + 2 * r] | ((unsigned)tok[8 + 2 * r + 1] << 16));
    }
    v8i gx = {};                                  // extra group (6-arg toolchain), zero-filled
    __builtin_amdgcn_tensor_load_to_lds(g0, g1, g2, g3, gx, 0);
    __builtin_amdgcn_s_wait_tensorcnt(0);
  }
  __syncthreads();

  const size_t wb = (size_t)e * II * DD + (size_t)(ibase + mn) * DD;
  const float* __restrict__ w1a = w1 + wb;
  const float* __restrict__ w1b = w1a + (size_t)16 * DD;
  const float* __restrict__ w3a = w3 + wb;
  const float* __restrict__ w3b = w3a + (size_t)16 * DD;
  const float* __restrict__ xr  = xs + mn * DD;

  v8f acc1a = {}, acc1b = {}, acc3a = {}, acc3b = {};
  for (int d0 = 0; d0 < DD; d0 += 32) {
    if (d0 + 32 < DD) {
      __builtin_prefetch(w1a + d0 + 32, 0, 1);
      __builtin_prefetch(w3a + d0 + 32, 0, 1);
    }
    v16bf a, b1a, b1b, b3a, b3b;
#pragma unroll
    for (int j = 0; j < 8; ++j) {
      const int kk = ((j < 4) ? (2 * j) : (16 + 2 * (j - 4))) + 8 * half;
      float2 av  = *(const float2*)(xr  + d0 + kk);
      float2 v1a = *(const float2*)(w1a + d0 + kk);
      float2 v1b = *(const float2*)(w1b + d0 + kk);
      float2 v3a = *(const float2*)(w3a + d0 + kk);
      float2 v3b = *(const float2*)(w3b + d0 + kk);
      a  [2 * j] = (__bf16)av.x;   a  [2 * j + 1] = (__bf16)av.y;
      b1a[2 * j] = (__bf16)v1a.x;  b1a[2 * j + 1] = (__bf16)v1a.y;
      b1b[2 * j] = (__bf16)v1b.x;  b1b[2 * j + 1] = (__bf16)v1b.y;
      b3a[2 * j] = (__bf16)v3a.x;  b3a[2 * j + 1] = (__bf16)v3a.y;
      b3b[2 * j] = (__bf16)v3b.x;  b3b[2 * j + 1] = (__bf16)v3b.y;
    }
    acc1a = __builtin_amdgcn_wmma_f32_16x16x32_bf16(false, a, false, b1a, (short)0, acc1a, false, false);
    acc3a = __builtin_amdgcn_wmma_f32_16x16x32_bf16(false, a, false, b3a, (short)0, acc3a, false, false);
    acc1b = __builtin_amdgcn_wmma_f32_16x16x32_bf16(false, a, false, b1b, (short)0, acc1b, false, false);
    acc3b = __builtin_amdgcn_wmma_f32_16x16x32_bf16(false, a, false, b3b, (short)0, acc3b, false, false);
  }
#pragma unroll
  for (int rr = 0; rr < 8; ++rr) {
    const int row = row0 + rr + 8 * half;
    if (row < n) {
      const int p2 = list[e * MAXP + row];
      __bf16* hp = hbuf + (size_t)p2 * II + ibase + mn;
      hp[0]  = (__bf16)(fast_silu(acc1a[rr]) * acc3a[rr]);
      hp[16] = (__bf16)(fast_silu(acc1b[rr]) * acc3b[rr]);
    }
  }
}

// ---------------- down projection: out = h W2 ----------------
// Each wave: 16(token) x 32(D) tile, two WMMAs sharing the bf16 A (h) operand.
// Grid: x = token-blocks (fast) for L2-temporal reuse of the W2 slice.

__global__ void __launch_bounds__(128)
k_down(const __bf16* __restrict__ hbuf, const float* __restrict__ w2,
       const int* __restrict__ cnt, const int* __restrict__ list,
       float* __restrict__ out)
{
  const int e    = blockIdx.z;
  const int n    = cnt[e];
  const int row0 = blockIdx.x * 16;
  if (row0 >= n) return;
  const int wave  = threadIdx.x >> 5;
  const int lane  = threadIdx.x & 31;
  const int half  = lane >> 4;
  const int mn    = lane & 15;
  const int dbase = (blockIdx.y * 8 + wave * 2) * 16;

  int r = row0 + mn; if (r > n - 1) r = n - 1;
  const int p = list[e * MAXP + r];
  const __bf16* __restrict__ hrow = hbuf + (size_t)p * II;
  const float*  __restrict__ w2a  = w2 + (size_t)e * II * DD + dbase + mn;
  const float*  __restrict__ w2b  = w2a + 16;

  v8f acca = {}, accb = {};
  for (int i0 = 0; i0 < II; i0 += 32) {
    v16bf a, ba, bb;
#pragma unroll
    for (int j = 0; j < 8; ++j) {
      const int kk = ((j < 4) ? (2 * j) : (16 + 2 * (j - 4))) + 8 * half;
      a [2 * j]     = hrow[i0 + kk];
      a [2 * j + 1] = hrow[i0 + kk + 1];
      ba[2 * j]     = (__bf16)w2a[(size_t)(i0 + kk)     * DD];
      ba[2 * j + 1] = (__bf16)w2a[(size_t)(i0 + kk + 1) * DD];
      bb[2 * j]     = (__bf16)w2b[(size_t)(i0 + kk)     * DD];
      bb[2 * j + 1] = (__bf16)w2b[(size_t)(i0 + kk + 1) * DD];
    }
    acca = __builtin_amdgcn_wmma_f32_16x16x32_bf16(false, a, false, ba, (short)0, acca, false, false);
    accb = __builtin_amdgcn_wmma_f32_16x16x32_bf16(false, a, false, bb, (short)0, accb, false, false);
  }
#pragma unroll
  for (int rr = 0; rr < 8; ++rr) {
    const int row = row0 + rr + 8 * half;
    if (row < n) {
      const int p2 = list[e * MAXP + row];
      float* op = out + (size_t)p2 * DD + dbase + mn;
      op[0]  = acca[rr];
      op[16] = accb[rr];
    }
  }
}

// ---------------- launcher ----------------

extern "C" void kernel_launch(void* const* d_in, const int* in_sizes, int n_in,
                              void* d_out, int out_size, void* d_ws, size_t ws_size,
                              hipStream_t stream) {
  (void)in_sizes; (void)n_in; (void)out_size; (void)ws_size;
  const float* x   = (const float*)d_in[0];
  const int*   idx = (const int*)d_in[1];
  const float* w1  = (const float*)d_in[2];
  const float* w2  = (const float*)d_in[3];
  const float* w3  = (const float*)d_in[4];
  float* out = (float*)d_out;

  // workspace: [16 int counters][E*MAXP int lists][MAXP*II bf16 hidden] ~23 MB
  int* cnt  = (int*)d_ws;
  int* list = cnt + 16;
  __bf16* hbuf = (__bf16*)(list + EE * MAXP);

  k_zero <<<1, 32, 0, stream>>>(cnt);
  k_route<<<(MAXP + 255) / 256, 256, 0, stream>>>(idx, cnt, list);
  // grid.x = token-blocks (fast dim) -> weight-slice L2 reuse
  k_up   <<<dim3(MAXP / 16, II / 128, EE), 128, 16 * DD * sizeof(float), stream>>>
         (x, w1, w3, cnt, list, hbuf);
  k_down <<<dim3(MAXP / 16, DD / 128, EE), 128, 0, stream>>>(hbuf, w2, cnt, list, out);
}
